// Head_72816875536420
// MI455X (gfx1250) — matrix operand order repaired and checked by reading, not compiled, not run on hardware
//
#include <hip/hip_runtime.h>
#include <hip/hip_bf16.h>

#define BATCH 8
#define SEQT  4096
#define CDIM  1024
#define HSZ   128
#define HHALF 2048                   // BLOCK/2
#define SCALE 0.08838834764831845f   // 1/sqrt(128)

typedef __attribute__((ext_vector_type(8)))  float        v8f;
typedef __attribute__((ext_vector_type(16))) __bf16       v16bf;
typedef __attribute__((ext_vector_type(4)))  unsigned int v4u;

union BF16x16 { v16bf v; v4u q[2]; };

__device__ __forceinline__ __bf16 tobf(float f) { return (__bf16)f; }  // native v_cvt (RNE)

// ---------------------------------------------------------------------------
// Kernel 1: q/k/v = x @ {Wq,Wk,Wv}, f32 in -> bf16 out via WMMA bf16 f32-acc.
// q,k stored row-major [B][T][HS]; v stored TRANSPOSED [B][HS][T] so the
// attention kernel can async-copy contiguous vT rows straight into LDS.
// grid.x = B*T/16 ; block = 256 (8 waves; wave w owns output cols [16w,16w+16)).
// ---------------------------------------------------------------------------
__global__ __launch_bounds__(256) void proj_kernel(
    const float* __restrict__ x,
    const float* __restrict__ Wq, const float* __restrict__ Wk,
    const float* __restrict__ Wv,
    __bf16* __restrict__ qb, __bf16* __restrict__ kb, __bf16* __restrict__ vb)
{
    __shared__ __bf16 sWT[128][136];   // W chunk transposed [n][k]; 272B rows (16B mult)

    const int blk  = blockIdx.x;
    const int b    = blk / (SEQT / 16);
    const int trow = (blk % (SEQT / 16)) * 16;
    const int wave = threadIdx.x >> 5;
    const int lane = threadIdx.x & 31;
    const int ln   = lane & 15;        // M for A-frag / N for B,D frags
    const int hs   = lane >> 4;        // half-select per ISA 16-bit layouts

    const float* xr = x + ((size_t)b * SEQT + trow + ln) * CDIM;

    for (int p = 0; p < 3; ++p) {
        const float* W = (p == 0) ? Wq : (p == 1) ? Wk : Wv;

        v8f acc = {0.f,0.f,0.f,0.f,0.f,0.f,0.f,0.f};

        for (int kc = 0; kc < CDIM / 128; ++kc) {
            __syncthreads();
            for (int i = threadIdx.x; i < 128 * 128; i += 256) {   // coalesced over n
                int k = i >> 7, n = i & 127;
                sWT[n][k] = tobf(W[(size_t)(kc * 128 + k) * HSZ + n]);
            }
            __syncthreads();

#pragma unroll
            for (int ks = 0; ks < 4; ++ks) {
                // A-frag (16x32 bf16): row ln; K runs [base,base+8) and [base+16,base+24)
                const int base = kc * 128 + ks * 32 + hs * 8;
                BF16x16 a;
#pragma unroll
                for (int e = 0; e < 8; ++e) {
                    a.v[e]     = tobf(xr[base + e]);
                    a.v[8 + e] = tobf(xr[base + 16 + e]);
                }
                // B-frag (32x16 bf16): col n = wave*16+ln, K runs [hs*8,+8) and [16+hs*8,+8)
                BF16x16 bb;
                const v4u* pp = reinterpret_cast<const v4u*>(
                    &sWT[wave * 16 + ln][ks * 32 + hs * 8]);
                bb.q[0] = pp[0]; bb.q[1] = pp[2];

                acc = __builtin_amdgcn_wmma_f32_16x16x32_bf16(
                    false, a.v, false, bb.v, (short)0, acc, false, false);
            }
        }
        // D layout: elem e -> row trow + e + hs*8, col wave*16 + ln
        if (p == 2) {  // v: store transposed [B][HS][T]
#pragma unroll
            for (int e = 0; e < 8; ++e)
                vb[((size_t)b * HSZ + wave * 16 + ln) * SEQT + trow + e + hs * 8] = tobf(acc[e]);
        } else {
            __bf16* dst = (p == 0) ? qb : kb;
#pragma unroll
            for (int e = 0; e < 8; ++e)
                dst[((size_t)b * SEQT + trow + e + hs * 8) * HSZ + wave * 16 + ln] = tobf(acc[e]);
        }
    }
}

// ---------------------------------------------------------------------------
// Kernel 2: banded flash attention. grid.x = B*(T/128); block = 256 (8 waves,
// wave w owns query rows [R+16w, R+16w+16)). Strips of 32 keys staged into LDS
// with GLOBAL_LOAD_ASYNC_TO_LDS_B128 (ASYNCcnt), block-uniform band range.
// ---------------------------------------------------------------------------
__global__ __launch_bounds__(256) void attn_kernel(
    const __bf16* __restrict__ qb,
    const __bf16* __restrict__ kb,
    const __bf16* __restrict__ vb,   // [B][HS][T]
    float* __restrict__ out)
{
    __shared__ __bf16 sK[32][128];    // k strip, row-major (256B rows)
    __shared__ __bf16 sVT[128][40];   // vT strip [d][s]; 80B rows (16B mult)
    __shared__ __bf16 sP[8][16][32];  // per-wave P relayout scratch (64B rows)

    const int b    = blockIdx.x >> 5;          // T/128 == 32 row-tiles
    const int R    = (blockIdx.x & 31) * 128;
    const int wave = threadIdx.x >> 5;
    const int lane = threadIdx.x & 31;
    const int ln   = lane & 15;
    const int hs   = lane >> 4;
    const int qrow0 = R + wave * 16;

    // LDS byte offsets (flat-LDS aperture keeps offset in addr[31:0])
    const unsigned ldsK = (unsigned)(uintptr_t)(&sK[0][0]);
    const unsigned ldsV = (unsigned)(uintptr_t)(&sVT[0][0]);

    // --- q fragments for this wave's 16 rows (kept in VGPRs whole kernel) ---
    v16bf qf[4];
    {
        const __bf16* qr = qb + ((size_t)b * SEQT + qrow0 + ln) * HSZ;
#pragma unroll
        for (int dc = 0; dc < 4; ++dc) {
            const int base = dc * 32 + hs * 8;
            BF16x16 t;
            t.q[0] = *reinterpret_cast<const v4u*>(qr + base);
            t.q[1] = *reinterpret_cast<const v4u*>(qr + base + 16);
            qf[dc] = t.v;
        }
    }

    // --- band bounds per owned row (pyr mask => contiguous [lo,hi]) ---
    int loE[8], hiE[8];
#pragma unroll
    for (int e = 0; e < 8; ++e) {
        int r = qrow0 + e + hs * 8;
        loE[e] = HHALF - 1 - (r >> 1);
        hiE[e] = r ? (HHALF + ((r - 1) >> 1)) : (HHALF - 1);
    }

    float mrow[8], lrow[8];
    v8f oacc[8];
    const v8f vzero = {0.f,0.f,0.f,0.f,0.f,0.f,0.f,0.f};
#pragma unroll
    for (int e = 0; e < 8; ++e) { mrow[e] = -1e30f; lrow[e] = 0.f; }
#pragma unroll
    for (int nt = 0; nt < 8; ++nt) oacc[nt] = vzero;

    const int rmax   = R + 127;
    const int lo_blk = (HHALF - 1 - (rmax >> 1)) & ~31;
    const int hi_blk = HHALF + ((rmax - 1) >> 1);

    for (int s0 = lo_blk; s0 <= hi_blk; s0 += 32) {
        // ---- async stage: k strip (8KB, verbatim) ----
        {
            const __bf16* ksrc = kb + ((size_t)b * SEQT + s0) * HSZ;
            for (int i = threadIdx.x; i < 512; i += 256) {
                unsigned loff = ldsK + (unsigned)i * 16u;
                unsigned goff = (unsigned)i * 16u;
                asm volatile("global_load_async_to_lds_b128 %0, %1, %2 offset:0"
                             :: "v"(loff), "v"(goff), "s"(ksrc) : "memory");
            }
        }
        // ---- async stage: vT strip (128 rows x 64B contiguous) ----
        {
            const __bf16* vsrc = vb + (size_t)b * HSZ * SEQT + s0;
            for (int i = threadIdx.x; i < 512; i += 256) {
                int d = i >> 2, part = i & 3;
                unsigned loff = ldsV + (unsigned)(d * 40 + part * 8) * 2u;
                unsigned goff = (unsigned)(d * SEQT + part * 8) * 2u;
                asm volatile("global_load_async_to_lds_b128 %0, %1, %2 offset:0"
                             :: "v"(loff), "v"(goff), "s"(vsrc) : "memory");
            }
        }
        if (s0 + 32 <= hi_blk)  // prefetch next k strip (global_prefetch_b8)
            __builtin_prefetch(kb + ((size_t)b * SEQT + s0 + 32) * HSZ + threadIdx.x * 16, 0, 1);
        asm volatile("s_wait_asynccnt 0" ::: "memory");
        __syncthreads();

        // ---- scores: two 16x16 tiles over 4 d-chunks each ----
        float pr[2][8];
#pragma unroll
        for (int j = 0; j < 2; ++j) {
            v8f sacc = vzero;
#pragma unroll
            for (int dc = 0; dc < 4; ++dc) {
                BF16x16 bk;  // B-frag: col = key j*16+ln, K = d runs dc*32 + [hs*8,+8) and [16+hs*8,+8)
                const v4u* pp = reinterpret_cast<const v4u*>(
                    &sK[j * 16 + ln][dc * 32 + hs * 8]);
                bk.q[0] = pp[0]; bk.q[1] = pp[2];
                sacc = __builtin_amdgcn_wmma_f32_16x16x32_bf16(
                    false, qf[dc], false, bk.v, (short)0, sacc, false, false);
            }
#pragma unroll
            for (int e = 0; e < 8; ++e) {
                int c = s0 + j * 16 + ln;
                float sv = sacc[e] * SCALE;
                pr[j][e] = (c >= loE[e] && c <= hiE[e]) ? sv : -1e30f;
            }
        }

        // ---- online softmax (rows live across the 16 lanes of each half) ----
        float scl[8];
#pragma unroll
        for (int e = 0; e < 8; ++e) {
            float mx = fmaxf(pr[0][e], pr[1][e]);
#pragma unroll
            for (int off = 1; off < 16; off <<= 1)
                mx = fmaxf(mx, __shfl_xor(mx, off, 32));
            float mn = fmaxf(mrow[e], mx);
            float s  = __expf(mrow[e] - mn);
            float p0 = __expf(pr[0][e] - mn);
            float p1 = __expf(pr[1][e] - mn);
            float rs = p0 + p1;
#pragma unroll
            for (int off = 1; off < 16; off <<= 1)
                rs += __shfl_xor(rs, off, 32);
            lrow[e] = lrow[e] * s + rs;
            mrow[e] = mn;
            scl[e]  = s;
            pr[0][e] = p0; pr[1][e] = p1;
        }
#pragma unroll
        for (int nt = 0; nt < 8; ++nt)
#pragma unroll
            for (int e = 0; e < 8; ++e) oacc[nt][e] *= scl[e];

        // ---- relayout P (D-layout) -> A-fragment via per-wave LDS scratch ----
#pragma unroll
        for (int e = 0; e < 8; ++e) {
            sP[wave][e + hs * 8][ln]      = tobf(pr[0][e]);
            sP[wave][e + hs * 8][16 + ln] = tobf(pr[1][e]);
        }
        __builtin_amdgcn_wave_barrier();
        BF16x16 pa;  // A-frag 16x32: row = ln, K runs [hs*8,+8) and [16+hs*8,+8)
        pa.q[0] = *reinterpret_cast<const v4u*>(&sP[wave][ln][hs * 8]);
        pa.q[1] = *reinterpret_cast<const v4u*>(&sP[wave][ln][16 + hs * 8]);

        // ---- PV: 8 N-tiles of the 16x128 output accumulator ----
#pragma unroll
        for (int nt = 0; nt < 8; ++nt) {
            BF16x16 bv;  // B-frag: col n = nt*16+ln, K = s runs [hs*8,+8) and [16+hs*8,+8)
            const v4u* pv = reinterpret_cast<const v4u*>(&sVT[nt * 16 + ln][hs * 8]);
            bv.q[0] = pv[0]; bv.q[1] = pv[2];
            oacc[nt] = __builtin_amdgcn_wmma_f32_16x16x32_bf16(
                false, pa.v, false, bv.v, (short)0, oacc[nt], false, false);
        }
        __syncthreads();
    }

    // ---- epilogue: out = acc / l ----
#pragma unroll
    for (int nt = 0; nt < 8; ++nt)
#pragma unroll
        for (int e = 0; e < 8; ++e) {
            size_t row = (size_t)b * SEQT + qrow0 + e + hs * 8;
            out[row * HSZ + nt * 16 + ln] = oacc[nt][e] / lrow[e];
        }
}

// ---------------------------------------------------------------------------
extern "C" void kernel_launch(void* const* d_in, const int* in_sizes, int n_in,
                              void* d_out, int out_size, void* d_ws, size_t ws_size,
                              hipStream_t stream) {
    const float* x  = (const float*)d_in[0];
    const float* Wq = (const float*)d_in[1];
    const float* Wk = (const float*)d_in[2];
    const float* Wv = (const float*)d_in[3];
    float* out = (float*)d_out;

    const size_t elems = (size_t)BATCH * SEQT * HSZ;   // 4 Mi bf16 per tensor
    __bf16* qb = (__bf16*)d_ws;
    __bf16* kb = qb + elems;
    __bf16* vb = kb + elems;

    proj_kernel<<<dim3((BATCH * SEQT) / 16), 256, 0, stream>>>(x, Wq, Wk, Wv, qb, kb, vb);
    attn_kernel<<<dim3(BATCH * (SEQT / 128)), 256, 0, stream>>>(qb, kb, vb, out);
}
